// CosSimRouter_learn_49778670960796
// MI455X (gfx1250) — matrix operand order repaired and checked by reading, not compiled, not run on hardware
//
#include <hip/hip_runtime.h>
#include <hip/hip_bf16.h>
#include <math.h>

#define E     4096
#define NH    16
#define DH    256
#define HIDD  8192
#define LVV   576
#define LTT   128
#define MC    704   /* padded cat rows  (LVV+LTT), 44 tiles */
#define MR    576   /* padded rem rows, 36 tiles */
#define MT_C  44
#define MT_R  36
#define GAMMA_C 0.2f
#define TEMP_C  0.05f

typedef __bf16 bf16;
typedef __attribute__((ext_vector_type(16))) bf16  v16bf;
typedef __attribute__((ext_vector_type(8)))  float v8f;
typedef unsigned short u16;
typedef unsigned int   u32;

__device__ __forceinline__ u16 f2bf(float f) {
    union { float f; u32 u; } a; a.f = f;
    u32 r = a.u + 0x7FFFu + ((a.u >> 16) & 1u);
    return (u16)(r >> 16);
}

__device__ __forceinline__ float gelu_f(float x) {
    return 0.5f * x * (1.0f + erff(x * 0.70710678118654752f));
}

union FragU {
    v16bf v;
    u16   s[16];
    uint4 q[2];
};

// A fragment (16x32 bf16): lane holds row l%16; elems 0..15 = K kt+16h..kt+16h+15,
// with k0 = kt + ((lane&16)?8:0)
__device__ __forceinline__ v16bf load_afrag(const u16* rowbase, int k0) {
    FragU f;
    f.q[0] = *(const uint4*)(rowbase + k0 + (k0 & 8));
    f.q[1] = *(const uint4*)(rowbase + k0 + (k0 & 8) + 8);
    return f.v;
}

// B fragment (32x16 bf16): lane holds col l%16; elems 0..15 = K kb..kb+15,
// with kb = kt + ((lane&16)?16:0)
__device__ __forceinline__ v16bf load_bfrag_bf16(const u16* rowbase, int kb) {
    FragU f;
    f.q[0] = *(const uint4*)(rowbase + kb);
    f.q[1] = *(const uint4*)(rowbase + kb + 8);
    return f.v;
}

__device__ __forceinline__ v16bf load_bfrag_f32(const float* rowbase, int kb) {
    FragU f;
    const float4* p = (const float4*)(rowbase + kb);
    float4 a = p[0], b = p[1], c = p[2], d = p[3];
    f.s[0]  = f2bf(a.x); f.s[1]  = f2bf(a.y); f.s[2]  = f2bf(a.z); f.s[3]  = f2bf(a.w);
    f.s[4]  = f2bf(b.x); f.s[5]  = f2bf(b.y); f.s[6]  = f2bf(b.z); f.s[7]  = f2bf(b.w);
    f.s[8]  = f2bf(c.x); f.s[9]  = f2bf(c.y); f.s[10] = f2bf(c.z); f.s[11] = f2bf(c.w);
    f.s[12] = f2bf(d.x); f.s[13] = f2bf(d.y); f.s[14] = f2bf(d.z); f.s[15] = f2bf(d.w);
    return f.v;
}

// ---------------------------------------------------------------------------
// Weight-streaming GEMM: C[M,N] = act(A[M,K] @ B[N,K]^T + bias)
// Grid = N/16 blocks of 256 threads (8 waves). Each block owns one 16-col
// N-tile and loops the full K; each wave accumulates up to 6 M-tiles in
// registers so every B (weight) element is read from HBM exactly once.
// ---------------------------------------------------------------------------
template<bool BF16B>
__global__ void __launch_bounds__(256)
gemm_wmma(const u16* __restrict__ A, int strideA, size_t aHeadStride, int nPerHead,
          const void* __restrict__ B, int strideB,
          const float* __restrict__ bias,
          float* __restrict__ Cf, u16* __restrict__ Cb, int strideC,
          int Mtiles, int K, int act)
{
    int lane = threadIdx.x & 31;
    int wave = threadIdx.x >> 5;
    int n0   = blockIdx.x << 4;
    if (nPerHead > 0) A += (size_t)(n0 / nPerHead) * aHeadStride;
    int nl = n0 + (lane & 15);
    const float* Browf = (const float*)B + (size_t)nl * strideB;
    const u16*   Browh = (const u16*)B + (size_t)nl * strideB;
    int ka = (lane & 16) ? 8 : 0;
    int kb = (lane & 16) ? 16 : 0;
    int am = lane & 15;

    v8f z = {0.f,0.f,0.f,0.f,0.f,0.f,0.f,0.f};
    v8f acc0=z, acc1=z, acc2=z, acc3=z, acc4=z, acc5=z;

    for (int kt = 0; kt < K; kt += 32) {
        // stream-prefetch the weight row ~8 K-steps ahead (global_prefetch_b8)
        if (kt + 256 < K) {
            if (BF16B) __builtin_prefetch(Browh + kt + 256, 0, 3);
            else       __builtin_prefetch(Browf + kt + 256, 0, 3);
        }
        v16bf bf_ = BF16B ? load_bfrag_bf16(Browh, kt + kb)
                          : load_bfrag_f32(Browf, kt + kb);
#define DO_MT(IDX, ACC)                                                          \
        { int mt = wave + (IDX) * 8;                                             \
          if (mt < Mtiles) {                                                     \
            v16bf af = load_afrag(A + (size_t)(mt * 16 + am) * strideA, kt + ka);\
            ACC = __builtin_amdgcn_wmma_f32_16x16x32_bf16(false, af, false, bf_, \
                                                          (short)0, ACC, false, false); } }
        DO_MT(0, acc0) DO_MT(1, acc1) DO_MT(2, acc2)
        DO_MT(3, acc3) DO_MT(4, acc4) DO_MT(5, acc5)
#undef DO_MT
    }

    float bb  = bias ? bias[nl] : 0.f;
    int rofs  = (lane >> 4) << 3;
#define ST_MT(IDX, ACC)                                                          \
    { int mt = wave + (IDX) * 8;                                                 \
      if (mt < Mtiles) {                                                         \
        union { v8f v; float f[8]; } u; u.v = ACC;                               \
        for (int vv = 0; vv < 8; vv++) {                                         \
            float x = u.f[vv] + bb;                                              \
            if (act == 1) x = gelu_f(x);                                         \
            size_t off = (size_t)(mt * 16 + rofs + vv) * strideC + nl;           \
            if (Cf) Cf[off] = x;                                                 \
            if (Cb) Cb[off] = f2bf(x);                                           \
        } } }
    ST_MT(0, acc0) ST_MT(1, acc1) ST_MT(2, acc2)
    ST_MT(3, acc3) ST_MT(4, acc4) ST_MT(5, acc5)
#undef ST_MT
}

// ---------------------------------------------------------------------------
// Q @ K^T per head: att[h][q][k] = scale * sum_d Q[q, h*DH+d] * K[k, h*DH+d]
// ---------------------------------------------------------------------------
__global__ void __launch_bounds__(256)
qk_wmma(const u16* __restrict__ Qb, int strideQ,
        const u16* __restrict__ Kb, int strideK,
        float* __restrict__ att, int Mq, int Mk, float scale)
{
    int lane = threadIdx.x & 31;
    int wave = threadIdx.x >> 5;
    int KT   = Mk >> 4;
    int KT8  = (KT + 7) >> 3;
    int b    = blockIdx.x;
    int k8   = b % KT8; b /= KT8;
    int QT   = Mq >> 4;
    int qt   = b % QT;
    int h    = b / QT;
    int kt   = k8 * 8 + wave;
    if (kt >= KT) return;

    int ka = (lane & 16) ? 8 : 0;
    int kb = (lane & 16) ? 16 : 0;
    const u16* qrow = Qb + (size_t)(qt * 16 + (lane & 15)) * strideQ + h * DH;
    const u16* krow = Kb + (size_t)(kt * 16 + (lane & 15)) * strideK + h * DH;

    v8f acc = {0.f,0.f,0.f,0.f,0.f,0.f,0.f,0.f};
#pragma unroll
    for (int d = 0; d < DH; d += 32) {
        v16bf af  = load_afrag(qrow, d + ka);
        v16bf bf_ = load_bfrag_bf16(krow, d + kb);
        acc = __builtin_amdgcn_wmma_f32_16x16x32_bf16(false, af, false, bf_,
                                                      (short)0, acc, false, false);
    }
    union { v8f v; float f[8]; } u; u.v = acc;
    int col   = kt * 16 + (lane & 15);
    int rbase = qt * 16 + ((lane >> 4) << 3);
#pragma unroll
    for (int vv = 0; vv < 8; vv++)
        att[((size_t)h * Mq + (rbase + vv)) * Mk + col] = u.f[vv] * scale;
}

// masked softmax over keys; rows >= validQ -> zeros; cols >= validK -> 0
__global__ void __launch_bounds__(256)
att_softmax(const float* __restrict__ att, u16* __restrict__ attb,
            int Mq, int Mk, const int* __restrict__ meta, int vq, int vk)
{
    __shared__ float sbuf[256];
    int r = blockIdx.x % Mq;
    int h = blockIdx.x / Mq;
    int tid = threadIdx.x;
    const float* row = att + ((size_t)h * Mq + r) * Mk;
    u16* brow = attb + ((size_t)h * Mq + r) * Mk;
    int validQ = meta[vq], validK = meta[vk];
    if (r >= validQ) {
        for (int c = tid; c < Mk; c += 256) brow[c] = 0;
        return;
    }
    float lv[3];           // Mk <= 704 -> at most 3 chunks per thread
    int   nch = 0;
    float m = -3.0e38f;
    for (int c = tid; c < validK; c += 256) {
        float x = row[c];
        lv[nch++] = x;
        m = fmaxf(m, x);
    }
    sbuf[tid] = m; __syncthreads();
    for (int s = 128; s > 0; s >>= 1) { if (tid < s) sbuf[tid] = fmaxf(sbuf[tid], sbuf[tid + s]); __syncthreads(); }
    m = sbuf[0]; __syncthreads();
    float sum = 0.f;
    for (int i = 0; i < nch; i++) { lv[i] = __expf(lv[i] - m); sum += lv[i]; }
    sbuf[tid] = sum; __syncthreads();
    for (int s = 128; s > 0; s >>= 1) { if (tid < s) sbuf[tid] += sbuf[tid + s]; __syncthreads(); }
    float inv = 1.0f / sbuf[0];
    int i = 0;
    for (int c = tid; c < Mk; c += 256) {
        float p = (c < validK) ? lv[i++] * inv : 0.f;
        brow[c] = f2bf(p);
    }
}

// out[c*M + r] = in[r*strideIn + c]  (V -> V^T for att@V B-operand)
__global__ void transpose_bf(const u16* __restrict__ in, int strideIn,
                             u16* __restrict__ out, int M)
{
    size_t i = (size_t)blockIdx.x * 256 + threadIdx.x;
    size_t total = (size_t)M * E;
    if (i >= total) return;
    int r = (int)(i / E), c = (int)(i % E);
    out[(size_t)c * M + r] = in[(size_t)r * strideIn + c];
}

// y = LN(a + b), row-wise; writes fp32 and optional bf16
__global__ void __launch_bounds__(256)
ln_residual(const float* __restrict__ a, const float* __restrict__ b,
            float* __restrict__ outf, u16* __restrict__ outb)
{
    __shared__ float s1[256], s2[256];
    int r = blockIdx.x, tid = threadIdx.x;
    const float* ra = a + (size_t)r * E;
    const float* rb = b + (size_t)r * E;
    float s = 0.f, ss = 0.f, vals[16];
#pragma unroll
    for (int j = 0; j < 16; j++) {
        float x = ra[tid + j * 256] + rb[tid + j * 256];
        vals[j] = x; s += x; ss += x * x;
    }
    s1[tid] = s; s2[tid] = ss; __syncthreads();
    for (int st = 128; st > 0; st >>= 1) {
        if (tid < st) { s1[tid] += s1[tid + st]; s2[tid] += s2[tid + st]; }
        __syncthreads();
    }
    float mean = s1[0] * (1.0f / E);
    float var  = s2[0] * (1.0f / E) - mean * mean;
    float rstd = rsqrtf(fmaxf(var, 0.f) + 1e-5f);
#pragma unroll
    for (int j = 0; j < 16; j++) {
        float y = (vals[j] - mean) * rstd;
        size_t off = (size_t)r * E + tid + j * 256;
        outf[off] = y;
        if (outb) outb[off] = f2bf(y);
    }
}

__global__ void __launch_bounds__(256)
row_invnorm(const float* __restrict__ vf, const float* __restrict__ te,
            float* __restrict__ inv_v, float* __restrict__ inv_t)
{
    __shared__ float sbuf[256];
    int r = blockIdx.x, tid = threadIdx.x;
    const float* p = (r < LVV) ? vf + (size_t)r * E : te + (size_t)(r - LVV) * E;
    float ss = 0.f;
    for (int c = tid; c < E; c += 256) { float x = p[c]; ss += x * x; }
    sbuf[tid] = ss; __syncthreads();
    for (int s = 128; s > 0; s >>= 1) { if (tid < s) sbuf[tid] += sbuf[tid + s]; __syncthreads(); }
    if (tid == 0) {
        float inv = 1.0f / fmaxf(sqrtf(sbuf[0]), 1e-8f);
        if (r < LVV) inv_v[r] = inv; else inv_t[r - LVV] = inv;
    }
}

__global__ void __launch_bounds__(128)
cos_max(const float* __restrict__ vf, const float* __restrict__ te,
        const float* __restrict__ inv_v, const float* __restrict__ inv_t,
        const unsigned char* __restrict__ mask, float* __restrict__ smax)
{
    __shared__ float vrow[E];
    __shared__ float red[128];
    int i = blockIdx.x, tid = threadIdx.x;
    for (int c = tid * 4; c < E; c += 128 * 4)
        *(float4*)&vrow[c] = *(const float4*)&vf[(size_t)i * E + c];
    __syncthreads();
    const float4* trow = (const float4*)(te + (size_t)tid * E);
    const float4* vr4  = (const float4*)vrow;
    float dot = 0.f;
    for (int c = 0; c < E / 4; c++) {
        float4 a = vr4[c], b = trow[c];
        dot += a.x * b.x + a.y * b.y + a.z * b.z + a.w * b.w;
    }
    float val = mask[tid] ? dot * inv_v[i] * inv_t[tid] : 0.f;
    red[tid] = val; __syncthreads();
    for (int s = 64; s > 0; s >>= 1) { if (tid < s) red[tid] = fmaxf(red[tid], red[tid + s]); __syncthreads(); }
    if (tid == 0) smax[i] = red[0];
}

// softmax over vision tokens, bitonic sort desc, cumsum threshold -> sel/rem
__global__ void __launch_bounds__(1024)
router_select(const float* __restrict__ smax, int* __restrict__ meta,
              int* __restrict__ sel_idx, int* __restrict__ rem_idx)
{
    __shared__ float v[1024]; __shared__ int id[1024];
    __shared__ float red[1024]; __shared__ int st;
    int tid = threadIdx.x;
    float l = (tid < LVV) ? smax[tid] * (1.0f / TEMP_C) : -3.0e38f;
    red[tid] = l; __syncthreads();
    for (int s = 512; s > 0; s >>= 1) { if (tid < s) red[tid] = fmaxf(red[tid], red[tid + s]); __syncthreads(); }
    float mx = red[0]; __syncthreads();
    float e = (tid < LVV) ? __expf(l - mx) : 0.f;
    red[tid] = e; __syncthreads();
    for (int s = 512; s > 0; s >>= 1) { if (tid < s) red[tid] += red[tid + s]; __syncthreads(); }
    float sum = red[0]; __syncthreads();
    v[tid] = (tid < LVV) ? e / sum : -1.0f;
    id[tid] = tid; __syncthreads();
    for (int k = 2; k <= 1024; k <<= 1)
        for (int j = k >> 1; j > 0; j >>= 1) {
            int ixj = tid ^ j;
            if (ixj > tid) {
                bool seg = (tid & k) == 0;
                bool sw = seg ? (v[tid] < v[ixj]) : (v[tid] > v[ixj]);
                if (sw) {
                    float tv = v[tid]; v[tid] = v[ixj]; v[ixj] = tv;
                    int ti = id[tid]; id[tid] = id[ixj]; id[ixj] = ti;
                }
            }
            __syncthreads();
        }
    if (tid == 0) {
        float cum = 0.f; int t = 0;
        for (int i = 0; i < LVV; i++) { cum += v[i]; if (cum <= GAMMA_C) t++; }
        st = t;
        meta[0] = t; meta[1] = t + LTT; meta[2] = LVV - t;
        meta[3] = (int)((double)t * 0.7);
    }
    __syncthreads();
    int t = st;
    if (tid < t) sel_idx[tid] = id[tid];
    if (tid >= t && tid < LVV) rem_idx[tid - t] = id[tid];
}

__global__ void __launch_bounds__(256)
build_cat(const float* __restrict__ vf, const float* __restrict__ te,
          const int* __restrict__ meta, const int* __restrict__ sel_idx,
          float* __restrict__ catf, u16* __restrict__ catb)
{
    int r = blockIdx.x, t = meta[0];
    const float* src = nullptr;
    if (r < t) src = vf + (size_t)sel_idx[r] * E;
    else if (r < t + LTT) src = te + (size_t)(r - t) * E;
    for (int c = threadIdx.x; c < E; c += 256) {
        float x = src ? src[c] : 0.f;
        catf[(size_t)r * E + c] = x;
        catb[(size_t)r * E + c] = f2bf(x);
    }
}

__global__ void __launch_bounds__(256)
build_rem(const float* __restrict__ vf, const int* __restrict__ meta,
          const int* __restrict__ rem_idx, float* __restrict__ remf, u16* __restrict__ remb)
{
    int r = blockIdx.x, Lr = meta[2];
    const float* src = (r < Lr) ? vf + (size_t)rem_idx[r] * E : nullptr;
    for (int c = threadIdx.x; c < E; c += 256) {
        float x = src ? src[c] : 0.f;
        remf[(size_t)r * E + c] = x;
        remb[(size_t)r * E + c] = f2bf(x);
    }
}

__global__ void __launch_bounds__(256)
score_sigmoid(const float* __restrict__ x, const float* __restrict__ Ws,
              const float* __restrict__ bs, float* __restrict__ es)
{
    __shared__ float sbuf[256];
    int r = blockIdx.x, tid = threadIdx.x;
    const float* row = x + (size_t)r * E;
    float s = 0.f;
    for (int c = tid; c < E; c += 256) s += row[c] * Ws[c];
    sbuf[tid] = s; __syncthreads();
    for (int st = 128; st > 0; st >>= 1) { if (tid < st) sbuf[tid] += sbuf[tid + st]; __syncthreads(); }
    if (tid == 0) {
        float z = sbuf[0] + bs[0];
        es[r] = 1.0f / (1.0f + __expf(-z));
    }
}

// top-k of es over valid rem rows, merge with sel, sort ascending, emit final
__global__ void __launch_bounds__(1024)
topk_final(const float* __restrict__ es, const int* __restrict__ meta,
           const int* __restrict__ sel_idx, const int* __restrict__ rem_idx,
           int* __restrict__ final_idx, int out_rows)
{
    __shared__ float v[1024]; __shared__ int id[1024]; __shared__ int fin[1024];
    int tid = threadIdx.x;
    int t = meta[0], Lr = meta[2], kx = meta[3];
    v[tid] = (tid < Lr) ? es[tid] : -3.0e38f;
    id[tid] = tid; __syncthreads();
    for (int k = 2; k <= 1024; k <<= 1)
        for (int j = k >> 1; j > 0; j >>= 1) {
            int ixj = tid ^ j;
            if (ixj > tid) {
                bool seg = (tid & k) == 0;
                bool sw = seg ? (v[tid] < v[ixj]) : (v[tid] > v[ixj]);
                if (sw) {
                    float tv = v[tid]; v[tid] = v[ixj]; v[ixj] = tv;
                    int ti = id[tid]; id[tid] = id[ixj]; id[ixj] = ti;
                }
            }
            __syncthreads();
        }
    int n = t + kx; if (n > 1024) n = 1024;
    if (tid < t) fin[tid] = sel_idx[tid];
    else if (tid < n) fin[tid] = rem_idx[id[tid - t]];
    else fin[tid] = 0x7FFFFFFF;
    __syncthreads();
    for (int k = 2; k <= 1024; k <<= 1)
        for (int j = k >> 1; j > 0; j >>= 1) {
            int ixj = tid ^ j;
            if (ixj > tid) {
                bool seg = (tid & k) == 0;
                bool sw = seg ? (fin[tid] > fin[ixj]) : (fin[tid] < fin[ixj]);
                if (sw) { int tv = fin[tid]; fin[tid] = fin[ixj]; fin[ixj] = tv; }
            }
            __syncthreads();
        }
    if (tid < out_rows) {
        int src = (tid < n) ? fin[tid] : fin[(n > 0) ? (n - 1) : 0];
        if (src < 0 || src >= LVV) src = 0;
        final_idx[tid] = src;
    }
}

__global__ void __launch_bounds__(256)
gather_rows(const float* __restrict__ vf, const int* __restrict__ final_idx,
            float* __restrict__ out)
{
    int r = blockIdx.x;
    int src = final_idx[r];
    if (src < 0 || src >= LVV) src = 0;
    const float* s = vf + (size_t)src * E;
    float* d = out + (size_t)r * E;
    for (int c = threadIdx.x; c < E; c += 256) d[c] = s[c];
}

// ---------------------------------------------------------------------------
extern "C" void kernel_launch(void* const* d_in, const int* in_sizes, int n_in,
                              void* d_out, int out_size, void* d_ws, size_t ws_size,
                              hipStream_t stream)
{
    const float* vf    = (const float*)d_in[0];
    const float* te    = (const float*)d_in[1];
    const unsigned char* mask = (const unsigned char*)d_in[2];
    const float* Wqkv1 = (const float*)d_in[3];
    const float* bqkv1 = (const float*)d_in[4];
    const float* Wo1   = (const float*)d_in[5];
    const float* bo1   = (const float*)d_in[6];
    const float* Wqkv2 = (const float*)d_in[7];
    const float* bqkv2 = (const float*)d_in[8];
    const float* Wo2   = (const float*)d_in[9];
    const float* bo2   = (const float*)d_in[10];
    const float* Wqkvc = (const float*)d_in[11];
    const float* bqkvc = (const float*)d_in[12];
    const float* Woc   = (const float*)d_in[13];
    const float* boc   = (const float*)d_in[14];
    const float* Wf1   = (const float*)d_in[15];
    const float* bf1v  = (const float*)d_in[16];
    const float* Wf2   = (const float*)d_in[17];
    const float* bf2v  = (const float*)d_in[18];
    const float* Wsp   = (const float*)d_in[19];
    const float* bsp   = (const float*)d_in[20];
    (void)in_sizes; (void)n_in; (void)ws_size;

    char* wsb = (char*)d_ws;
    size_t off = 0;
    auto alloc = [&](size_t bytes) -> void* {
        void* p = wsb + off;
        off += (bytes + 255) & ~(size_t)255;
        return p;
    };
    int*   meta    = (int*)  alloc(64);
    float* inv_v   = (float*)alloc(LVV * 4);
    float* inv_t   = (float*)alloc(LTT * 4);
    float* smax    = (float*)alloc(LVV * 4);
    int*   sel_idx = (int*)  alloc(LVV * 4);
    int*   rem_idx = (int*)  alloc(LVV * 4);
    int*   fin_idx = (int*)  alloc(1024 * 4);
    float* es      = (float*)alloc(MR * 4);
    float* cat_f   = (float*)alloc((size_t)MC * E * 4);
    u16*   cat_b   = (u16*)  alloc((size_t)MC * E * 2);
    float* rem_f   = (float*)alloc((size_t)MR * E * 4);
    u16*   rem_b   = (u16*)  alloc((size_t)MR * E * 2);
    u16*   qkv_b   = (u16*)  alloc((size_t)MC * 3 * E * 2);
    u16*   vt_b    = (u16*)  alloc((size_t)E * MC * 2);
    float* att_f   = (float*)alloc((size_t)NH * MC * MC * 4);
    u16*   att_b   = (u16*)  alloc((size_t)NH * MC * MC * 2);
    float* o_f     = (float*)alloc((size_t)MC * E * 4);
    u16*   o_b     = (u16*)  alloc((size_t)MC * E * 2);
    float* proj_f  = (float*)alloc((size_t)MC * E * 4);
    float* x1_f    = (float*)alloc((size_t)MC * E * 4);
    u16*   x1_b    = (u16*)  alloc((size_t)MC * E * 2);
    float* r_f     = (float*)alloc((size_t)MR * E * 4);
    u16*   r_b     = (u16*)  alloc((size_t)MR * E * 2);
    float* x2_f    = (float*)alloc((size_t)MR * E * 4);
    u16*   x2_b    = (u16*)  alloc((size_t)MR * E * 2);
    u16*   h1_b    = (u16*)  alloc((size_t)MR * HIDD * 2);

    const float SC = 0.0625f; // 1/sqrt(DH)

    // ---- routing ----
    row_invnorm<<<LVV + LTT, 256, 0, stream>>>(vf, te, inv_v, inv_t);
    cos_max<<<LVV, 128, 0, stream>>>(vf, te, inv_v, inv_t, mask, smax);
    router_select<<<1, 1024, 0, stream>>>(smax, meta, sel_idx, rem_idx);
    build_cat<<<MC, 256, 0, stream>>>(vf, te, meta, sel_idx, cat_f, cat_b);
    build_rem<<<MR, 256, 0, stream>>>(vf, meta, rem_idx, rem_f, rem_b);

    // ---- block 1: self-attention on cat (Mpad=704) ----
    gemm_wmma<false><<<(3 * E) / 16, 256, 0, stream>>>(cat_b, E, 0, 0, Wqkv1, E, bqkv1,
        nullptr, qkv_b, 3 * E, MT_C, E, 0);
    qk_wmma<<<NH * MT_C * ((MT_C + 7) / 8), 256, 0, stream>>>(qkv_b, 3 * E, qkv_b + E, 3 * E,
        att_f, MC, MC, SC);
    att_softmax<<<NH * MC, 256, 0, stream>>>(att_f, att_b, MC, MC, meta, 1, 1);
    transpose_bf<<<(int)(((size_t)MC * E + 255) / 256), 256, 0, stream>>>(qkv_b + 2 * E, 3 * E, vt_b, MC);
    gemm_wmma<true><<<E / 16, 256, 0, stream>>>(att_b, MC, (size_t)MC * MC, DH, vt_b, MC,
        nullptr, o_f, o_b, E, MT_C, MC, 0);
    gemm_wmma<false><<<E / 16, 256, 0, stream>>>(o_b, E, 0, 0, Wo1, E, bo1,
        proj_f, nullptr, E, MT_C, E, 0);
    ln_residual<<<MC, 256, 0, stream>>>(proj_f, cat_f, x1_f, x1_b);

    // ---- block 2: self-attention on rem (Mpad=576) ----
    gemm_wmma<false><<<(3 * E) / 16, 256, 0, stream>>>(rem_b, E, 0, 0, Wqkv2, E, bqkv2,
        nullptr, qkv_b, 3 * E, MT_R, E, 0);
    qk_wmma<<<NH * MT_R * ((MT_R + 7) / 8), 256, 0, stream>>>(qkv_b, 3 * E, qkv_b + E, 3 * E,
        att_f, MR, MR, SC);
    att_softmax<<<NH * MR, 256, 0, stream>>>(att_f, att_b, MR, MR, meta, 2, 2);
    transpose_bf<<<(int)(((size_t)MR * E + 255) / 256), 256, 0, stream>>>(qkv_b + 2 * E, 3 * E, vt_b, MR);
    gemm_wmma<true><<<E / 16, 256, 0, stream>>>(att_b, MR, (size_t)MR * MR, DH, vt_b, MR,
        nullptr, o_f, o_b, E, MT_R, MR, 0);
    gemm_wmma<false><<<E / 16, 256, 0, stream>>>(o_b, E, 0, 0, Wo2, E, bo2,
        proj_f, nullptr, E, MT_R, E, 0);
    ln_residual<<<MR, 256, 0, stream>>>(proj_f, rem_f, r_f, r_b);

    // ---- cross attention: q from r(576), kv from x1(704) ----
    u16* qc_b = qkv_b;                       // 576 x 4096
    u16* kv_b = qkv_b + (size_t)MR * E;      // 704 x 8192
    gemm_wmma<false><<<E / 16, 256, 0, stream>>>(r_b, E, 0, 0, Wqkvc, E, bqkvc,
        nullptr, qc_b, E, MT_R, E, 0);
    gemm_wmma<false><<<(2 * E) / 16, 256, 0, stream>>>(x1_b, E, 0, 0, Wqkvc + (size_t)E * E, E,
        bqkvc + E, nullptr, kv_b, 2 * E, MT_C, E, 0);
    qk_wmma<<<NH * MT_R * ((MT_C + 7) / 8), 256, 0, stream>>>(qc_b, E, kv_b, 2 * E,
        att_f, MR, MC, SC);
    att_softmax<<<NH * MR, 256, 0, stream>>>(att_f, att_b, MR, MC, meta, 2, 1);
    transpose_bf<<<(int)(((size_t)MC * E + 255) / 256), 256, 0, stream>>>(kv_b + E, 2 * E, vt_b, MC);
    gemm_wmma<true><<<E / 16, 256, 0, stream>>>(att_b, MC, (size_t)MR * MC, DH, vt_b, MC,
        nullptr, o_f, o_b, E, MT_R, MC, 0);
    gemm_wmma<false><<<E / 16, 256, 0, stream>>>(o_b, E, 0, 0, Woc, E, boc,
        proj_f, nullptr, E, MT_R, E, 0);
    ln_residual<<<MR, 256, 0, stream>>>(proj_f, r_f, x2_f, x2_b);

    // ---- FFN + final LN ----
    gemm_wmma<false><<<HIDD / 16, 256, 0, stream>>>(x2_b, E, 0, 0, Wf1, E, bf1v,
        nullptr, h1_b, HIDD, MT_R, E, 1);
    gemm_wmma<false><<<E / 16, 256, 0, stream>>>(h1_b, HIDD, 0, 0, Wf2, HIDD, bf2v,
        proj_f, nullptr, E, MT_R, HIDD, 0);
    ln_residual<<<MR, 256, 0, stream>>>(proj_f, x2_f, o_f, nullptr);

    // ---- score + expand + final gather ----
    score_sigmoid<<<MR, 256, 0, stream>>>(o_f, Wsp, bsp, es);
    int out_rows = out_size / E;
    if (out_rows < 1) out_rows = 1;
    if (out_rows > 1024) out_rows = 1024;
    topk_final<<<1, 1024, 0, stream>>>(es, meta, sel_idx, rem_idx, fin_idx, out_rows);
    gather_rows<<<out_rows, 256, 0, stream>>>(vf, fin_idx, (float*)d_out);
}
